// SelectiveSSM_12472585028139
// MI455X (gfx1250) — compile-verified, matrix operand order checked by reading, not run on hardware
//
#include <hip/hip_runtime.h>
#include <hip/hip_bf16.h>
#include <math.h>

// ---------------------------------------------------------------------------
// Selective SSM (Mamba block) forward for MI455X / gfx1250.
// GEMMs: bf16 WMMA 16x16x32, f32 accumulate, LDS double-buffered B staging
// via gfx1250 async global->LDS loads (ASYNCcnt), ds_load/WMMA interleave
// pinned with sched_group_barrier so WMMAs overlap LDS reads.
// Scan: register-resident state, LDS-staged shared per-step params.
// ---------------------------------------------------------------------------

typedef __bf16 bf16_t;
typedef __attribute__((ext_vector_type(16))) __bf16 v16bf;
typedef __attribute__((ext_vector_type(8)))  __bf16 v8bf;
typedef __attribute__((ext_vector_type(8)))  float  v8f;
typedef __attribute__((ext_vector_type(4)))  int    v4i;

#define BATCH 4
#define SEQ   2048
#define DM    1024
#define DST   16
#define DCONV 4
#define BL    (BATCH * SEQ)   // 8192 rows
#define E2    (2 * DM)        // 2048
#define NPAD  64              // x_proj 33 outputs padded to 64 for WMMA
#define KSTAGE 128            // K per LDS stage
#define BSTRIDE (KSTAGE + 8)  // LDS row stride (bf16): 272B, bank-spread

// ---- gfx1250 async global->LDS copy (guarded: falls back to sync copy) ----
// Builtin signature (from hipcc diagnostic): (int4 AS1*, int4 AS3*, imm, imm)
#if __has_builtin(__builtin_amdgcn_global_load_async_to_lds_b128)
#define HAVE_ASYNC_LDS 1
#define ASYNC_COPY16(dst, src)                                       \
  __builtin_amdgcn_global_load_async_to_lds_b128(                    \
      (__attribute__((address_space(1))) v4i*)(src),                 \
      (__attribute__((address_space(3))) v4i*)(dst), 0, 0)
#else
#define ASYNC_COPY16(dst, src) (*(v8bf*)(dst) = *(const v8bf*)(src))
#endif

#if __has_builtin(__builtin_amdgcn_s_wait_asynccnt)
#define WAIT_ASYNC(n) __builtin_amdgcn_s_wait_asynccnt(n)
#elif defined(HAVE_ASYNC_LDS)
#define WAIT_ASYNC(n) asm volatile("s_wait_asynccnt %0" ::"i"(n) : "memory")
#else
#define WAIT_ASYNC(n) ((void)0)
#endif

// ---- scheduling directives: pin DS-read / WMMA interleave -----------------
// mask 0x100 = DS read, 0x008 = MFMA/WMMA class (best-effort, compile-safe)
#if __has_builtin(__builtin_amdgcn_sched_group_barrier)
#define SGB(mask, n) __builtin_amdgcn_sched_group_barrier(mask, n, 0)
#else
#define SGB(mask, n) ((void)0)
#endif

// --------------------------- conversion kernels ----------------------------

__global__ void k_cvt_bf16(const float* __restrict__ s, bf16_t* __restrict__ d, int n) {
  int i = blockIdx.x * blockDim.x + threadIdx.x;
  if (i < n) d[i] = (bf16_t)s[i];
}

// x_proj_w is [33, 1024]; pad to [64, 1024] with zeros so N is a WMMA multiple.
__global__ void k_pad_xproj(const float* __restrict__ s, bf16_t* __restrict__ d) {
  int i = blockIdx.x * blockDim.x + threadIdx.x;  // over 64*1024
  int row = i >> 10, col = i & 1023;
  d[i] = (row < (2 * DST + 1)) ? (bf16_t)s[row * DM + col] : (bf16_t)0.0f;
}

// ------------------------------ WMMA GEMM ----------------------------------
// C[M,N] = A[M,K] * B[N,K]^T   (both operands K-major / row-major over K)
// 16-bit A-fragment layout (ISA 7.12.2): lane holds row M = lane&15,
// K-half = (lane>>4)*8; VGPRs 0-3 hold K=kb..kb+7, VGPRs 4-7 hold K=16+kb..+7.
__device__ __forceinline__ v16bf load_fragA(const bf16_t* __restrict__ base,
                                            int row0, int ld, int k0, int lane) {
  const int r  = lane & 15;
  const int kb = (lane >> 4) * 8;
  const bf16_t* p = base + (size_t)(row0 + r) * ld + k0 + kb;
  v8bf lo = *(const v8bf*)p;          // global_load_b128
  v8bf hi = *(const v8bf*)(p + 16);
  v16bf f;
#pragma unroll
  for (int i = 0; i < 8; ++i) { f[i] = lo[i]; f[i + 8] = hi[i]; }
  return f;
}

// B fragment from the LDS-staged 64xKSTAGE slab (row-major, stride BSTRIDE).
__device__ __forceinline__ v16bf load_fragB_lds(const bf16_t* bb, int j, int kk, int lane) {
  const int r  = (lane & 15) + j * 16;
  const int kb = ((lane >> 4) << 3) + kk;
  const bf16_t* p = bb + r * BSTRIDE + kb;
  v8bf lo = *(const v8bf*)p;          // ds_load_b128
  v8bf hi = *(const v8bf*)(p + 16);
  v16bf f;
#pragma unroll
  for (int i = 0; i < 8; ++i) { f[i] = lo[i]; f[i + 8] = hi[i]; }
  return f;
}

// Block: 256 threads = 8 waves.  Block tile: 128(M) x 64(N).  K staged in
// 128-wide double-buffered LDS slabs shared by all 8 waves.
template <int STORE_BF16>
__global__ void __launch_bounds__(256)
k_gemm_bf16(const bf16_t* __restrict__ A, const bf16_t* __restrict__ Bm,
            void* __restrict__ Cv, int M, int N, int K) {
  __shared__ bf16_t Bs[2][64 * BSTRIDE];

  const int lane = threadIdx.x & 31;
  const int wid  = threadIdx.x >> 5;
  const int tiles_n = N >> 6;
  const int tm = blockIdx.x / tiles_n;
  const int tn = blockIdx.x % tiles_n;
  const int m0 = tm * 128 + wid * 16;
  if (m0 >= M) return;

  // B staging: thread -> (row 0..63, 32-elem K chunk 0..3); 4 async b128 each.
  const int srow = threadIdx.x >> 2;
  const int skol = (threadIdx.x & 3) * 32;
  const bf16_t* bsrc = Bm + (size_t)(tn * 64 + srow) * K + skol;

  auto stage = [&](int ks, int buf) {
    const bf16_t* s = bsrc + ks;
    bf16_t* d = &Bs[buf][srow * BSTRIDE + skol];
#pragma unroll
    for (int c = 0; c < 4; ++c) ASYNC_COPY16(d + 8 * c, s + 8 * c);
  };

  v8f acc[4] = {};
  stage(0, 0);
  int buf = 0;
  for (int ks = 0; ks < K; ks += KSTAGE, buf ^= 1) {
    const bool more = (ks + KSTAGE) < K;
    if (more) stage(ks + KSTAGE, buf ^ 1);       // next stage in flight
    // All 4 A fragments issued before wait+barrier: latency hidden by sync.
    v16bf a[4];
#pragma unroll
    for (int q = 0; q < 4; ++q) a[q] = load_fragA(A, m0, K, ks + 32 * q, lane);
    if (more) WAIT_ASYNC(4);     // current stage landed; next 4 outstanding
    else      WAIT_ASYNC(0);
    __syncthreads();             // all waves' slab portions visible

    const bf16_t* bb = &Bs[buf][0];
    // Software pipeline: B frags of quarter q+1 load during quarter q WMMAs.
    v16bf bq[4];
#pragma unroll
    for (int j = 0; j < 4; ++j) bq[j] = load_fragB_lds(bb, j, 0, lane);
#pragma unroll
    for (int q = 0; q < 4; ++q) {
      v16bf bn[4];
      if (q < 3) {
#pragma unroll
        for (int j = 0; j < 4; ++j) bn[j] = load_fragB_lds(bb, j, 32 * (q + 1), lane);
      }
#pragma unroll
      for (int j = 0; j < 4; ++j)
        acc[j] = __builtin_amdgcn_wmma_f32_16x16x32_bf16(false, a[q], false, bq[j],
                                                         (short)0, acc[j], false, false);
      if (q < 3) {
#pragma unroll
        for (int j = 0; j < 4; ++j) bq[j] = bn[j];
      }
    }
    // Pin the schedule: DS reads lead WMMA by one K-quarter.
    SGB(0x100, 8);               // q0 loads
    SGB(0x100, 8);               // q1 loads
    SGB(0x008, 4);               // q0 WMMAs
    SGB(0x100, 8);               // q2 loads
    SGB(0x008, 4);               // q1 WMMAs
    SGB(0x100, 8);               // q3 loads
    SGB(0x008, 4);               // q2 WMMAs
    SGB(0x008, 4);               // q3 WMMAs
    __syncthreads();             // protect buf^1 from restaging WAR
  }

  // D layout (ISA 7.12.2): element (M = vgpr + 8*(lane>>4), N = lane&15)
  const int n  = lane & 15;
  const int mh = (lane >> 4) * 8;
#pragma unroll
  for (int j = 0; j < 4; ++j) {
#pragma unroll
    for (int v = 0; v < 8; ++v) {
      size_t idx = (size_t)(m0 + mh + v) * N + tn * 64 + j * 16 + n;
      if (STORE_BF16) ((bf16_t*)Cv)[idx] = (bf16_t)acc[j][v];
      else            ((float*)Cv)[idx]  = acc[j][v];
    }
  }
}

// --------------------- depthwise causal conv + SiLU ------------------------
// xz is [BL, 2048] bf16: cols [0,1024) = x_ssm, cols [1024,2048) = z.
__global__ void k_conv_silu(const bf16_t* __restrict__ xz,
                            const float* __restrict__ cw,  // [1024,4]
                            const float* __restrict__ cb,  // [1024]
                            float* __restrict__ xcf, bf16_t* __restrict__ xcb) {
  int i = blockIdx.x * blockDim.x + threadIdx.x;
  if (i >= BL * DM) return;
  int d = i & (DM - 1);
  int l = (i >> 10) & (SEQ - 1);
  int b = i >> 21;
  float acc = cb[d];
#pragma unroll
  for (int k = 0; k < DCONV; ++k) {
    int lk = l + k - (DCONV - 1);      // left-padded causal window
    if (lk >= 0)
      acc += (float)xz[((size_t)(b * SEQ + lk)) * E2 + d] * cw[d * DCONV + k];
  }
  float s = acc / (1.0f + __expf(-acc));   // silu
  xcf[i] = s;
  xcb[i] = (bf16_t)s;
}

// ----------------------------- selective scan ------------------------------
// One thread per (b,d) channel: h[16] in registers; B/C/delta for each step
// shared across all 1024 d-channels of a batch -> staged through LDS in
// 16-step chunks.  Fuses +D*x, silu(z) gating, and bf16 output for out_proj.
__global__ void __launch_bounds__(256)
k_scan(const float* __restrict__ xcf,     // [BL, 1024] conv output (f32)
       const float* __restrict__ sp,      // [BL, 64]   ssm params (B|C|delta)
       const float* __restrict__ A_log,   // [1024, 16]
       const float* __restrict__ Dp_,     // [1024]
       const bf16_t* __restrict__ xz,     // [BL, 2048] for z gate
       bf16_t* __restrict__ yb) {         // [BL, 1024] gated output (bf16)
  const int b = blockIdx.x >> 2;
  const int d = ((blockIdx.x & 3) << 8) + threadIdx.x;

  float A[DST], h[DST];
#pragma unroll
  for (int n = 0; n < DST; ++n) { A[n] = -__expf(A_log[d * DST + n]); h[n] = 0.0f; }
  const float Dp = Dp_[d];

  __shared__ float smp[16 * 33];  // 16 steps x (16 B | 16 C | 1 delta)

  for (int l0 = 0; l0 < SEQ; l0 += 16) {
    __syncthreads();
    for (int i = threadIdx.x; i < 16 * 33; i += 256) {
      int r = i / 33, c = i - r * 33;
      smp[i] = sp[((size_t)(b * SEQ + l0 + r)) * NPAD + c];
    }
    __syncthreads();
#pragma unroll 1
    for (int ll = 0; ll < 16; ++ll) {
      const int l = l0 + ll;
      const float xt  = xcf[((size_t)(b * SEQ + l)) * DM + d];
      const float dtp = smp[ll * 33 + 32];
      const float dt  = (dtp > 20.0f) ? dtp : log1pf(__expf(dtp));  // softplus
      float yv = 0.0f;
#pragma unroll
      for (int n = 0; n < DST; ++n) {
        float dA = __expf(dt * A[n]);
        h[n] = dA * h[n] + (dt * smp[ll * 33 + n]) * xt;
        yv  += h[n] * smp[ll * 33 + 16 + n];
      }
      yv += Dp * xt;
      float zf = (float)xz[((size_t)(b * SEQ + l)) * E2 + DM + d];
      float sz = zf / (1.0f + __expf(-zf));                          // silu(z)
      yb[((size_t)(b * SEQ + l)) * DM + d] = (bf16_t)(yv * sz);
    }
  }
}

// ------------------------------- launcher ----------------------------------

extern "C" void kernel_launch(void* const* d_in, const int* in_sizes, int n_in,
                              void* d_out, int out_size, void* d_ws, size_t ws_size,
                              hipStream_t stream) {
  const float* x       = (const float*)d_in[0];
  const float* in_w    = (const float*)d_in[1];
  const float* conv_w  = (const float*)d_in[2];
  const float* conv_b  = (const float*)d_in[3];
  const float* xproj_w = (const float*)d_in[4];
  const float* A_log   = (const float*)d_in[5];
  const float* D_param = (const float*)d_in[6];
  const float* out_w   = (const float*)d_in[7];

  char* ws = (char*)d_ws;
  size_t o = 0;
  auto take = [&](size_t bytes) {
    char* p = ws + o;
    o += (bytes + 255) & ~(size_t)255;
    return p;
  };
  bf16_t* xb  = (bf16_t*)take((size_t)BL * DM * 2);    // x as bf16
  bf16_t* wib = (bf16_t*)take((size_t)E2 * DM * 2);    // in_proj_w bf16
  bf16_t* wxp = (bf16_t*)take((size_t)NPAD * DM * 2);  // x_proj_w padded bf16
  bf16_t* wob = (bf16_t*)take((size_t)DM * DM * 2);    // out_proj_w bf16
  bf16_t* xzb = (bf16_t*)take((size_t)BL * E2 * 2);    // xz (x_ssm|z) bf16
  float*  xcf = (float*) take((size_t)BL * DM * 4);    // conv output f32
  bf16_t* xcb = (bf16_t*)take((size_t)BL * DM * 2);    // conv output bf16
  float*  spf = (float*) take((size_t)BL * NPAD * 4);  // ssm params f32
  bf16_t* yb  = (bf16_t*)take((size_t)BL * DM * 2);    // gated y bf16

  // 1) precision conversion of activations + weights
  k_cvt_bf16<<<(BL * DM + 255) / 256, 256, 0, stream>>>(x, xb, BL * DM);
  k_cvt_bf16<<<(E2 * DM + 255) / 256, 256, 0, stream>>>(in_w, wib, E2 * DM);
  k_pad_xproj<<<(NPAD * DM) / 256, 256, 0, stream>>>(xproj_w, wxp);
  k_cvt_bf16<<<(DM * DM + 255) / 256, 256, 0, stream>>>(out_w, wob, DM * DM);

  // 2) in_proj: xz[BL,2048] = x * in_proj_w^T   (store bf16, stays in L2)
  k_gemm_bf16<1><<<(BL / 128) * (E2 / 64), 256, 0, stream>>>(xb, wib, xzb, BL, E2, DM);

  // 3) depthwise causal conv + SiLU
  k_conv_silu<<<(BL * DM) / 256, 256, 0, stream>>>(xzb, conv_w, conv_b, xcf, xcb);

  // 4) x_proj: ssm_params[BL,64] = x_conv * x_proj_w_pad^T (f32)
  k_gemm_bf16<0><<<(BL / 128) * (NPAD / 64), 256, 0, stream>>>(xcb, wxp, spf, BL, NPAD, DM);

  // 5) selective scan + D skip + silu(z) gate -> y (bf16)
  k_scan<<<BATCH * (DM / 256), 256, 0, stream>>>(xcf, spf, A_log, D_param, xzb, yb);

  // 6) out_proj: out[BL,1024] = y * out_proj_w^T (f32 to d_out)
  k_gemm_bf16<0><<<(BL / 128) * (DM / 64), 256, 0, stream>>>(yb, wob, d_out, BL, DM, DM);
}
